// GPT_43980465111718
// MI455X (gfx1250) — matrix operand order repaired and checked by reading, not compiled, not run on hardware
//
#include <hip/hip_runtime.h>
#include <hip/hip_bf16.h>

// ---------------------------------------------------------------------------
// Model constants (match the reference)
// ---------------------------------------------------------------------------
#define BATCH   2
#define SEQ     2048
#define DMODEL  1024
#define FFDIM   4096
#define NHEAD   16
#define HDIM    64
#define VOCAB   32000
#define MTOK    (BATCH * SEQ)      // 4096 token rows

typedef __attribute__((ext_vector_type(16))) __bf16 v16bf;
typedef __attribute__((ext_vector_type(8)))  __bf16 v8bf;
typedef __attribute__((ext_vector_type(4)))  __bf16 v4bf;
typedef __attribute__((ext_vector_type(2)))  __bf16 v2bf;
typedef __attribute__((ext_vector_type(8)))  float  v8f;
typedef __attribute__((ext_vector_type(4)))  float  v4f;

// Native f32 -> bf16 conversion (hardware cvt on gfx1250)
__device__ __forceinline__ __bf16 f2bf(float f) { return (__bf16)f; }
__device__ __forceinline__ v4bf f2bf4(v4f v) {
    return __builtin_convertvector(v, v4bf);
}

// ---------------------------------------------------------------------------
// DPP16 cross-lane reductions within a 16-lane half (no LDS traffic).
// quad_perm(1,0,3,2)=0xB1 (xor1), quad_perm(2,3,0,1)=0x4E (xor2),
// row_half_mirror=0x141 (xor across 8), row_mirror=0x140 (xor across 16).
// ---------------------------------------------------------------------------
template <int CTRL>
__device__ __forceinline__ float dpp_movf(float x) {
    return __builtin_bit_cast(
        float, __builtin_amdgcn_update_dpp(0, __builtin_bit_cast(int, x),
                                           CTRL, 0xF, 0xF, true));
}
__device__ __forceinline__ float redmax16(float v) {
    v = fmaxf(v, dpp_movf<0xB1>(v));
    v = fmaxf(v, dpp_movf<0x4E>(v));
    v = fmaxf(v, dpp_movf<0x141>(v));
    v = fmaxf(v, dpp_movf<0x140>(v));
    return v;
}
__device__ __forceinline__ float redsum16(float v) {
    v += dpp_movf<0xB1>(v);
    v += dpp_movf<0x4E>(v);
    v += dpp_movf<0x141>(v);
    v += dpp_movf<0x140>(v);
    return v;
}

// Assemble a 16x32 bf16 A-fragment for one lane from an LDS row pointer.
// 16-bit A layout: lane half=lane/16 selects K run
//   e=0..15 -> K = half*16 + e (contiguous, matches B fragments)
__device__ __forceinline__ v16bf ld_a_frag(const __bf16* rowp, int half) {
    v8bf lo = *(const v8bf*)(rowp + half * 16);
    v8bf hi = *(const v8bf*)(rowp + half * 16 + 8);
    v16bf r;
#pragma unroll
    for (int e = 0; e < 8; ++e) { r[e] = lo[e]; r[e + 8] = hi[e]; }
    return r;
}

// ---------------------------------------------------------------------------
// Embedding: x[b,s,:] = wte[tok[b,s],:] + wpe[s,:]
// ---------------------------------------------------------------------------
__global__ __launch_bounds__(256) void embed_kernel(
    const int* __restrict__ tokens, const float* __restrict__ wte,
    const float* __restrict__ wpe, float* __restrict__ x) {
    int bs = blockIdx.x;               // 0..MTOK-1
    int s  = bs & (SEQ - 1);
    int tok = tokens[bs];
    const v4f* wt = (const v4f*)(wte + (size_t)tok * DMODEL);
    const v4f* wp = (const v4f*)(wpe + (size_t)s * DMODEL);
    v4f* xo = (v4f*)(x + (size_t)bs * DMODEL);
    int i = threadIdx.x;               // DMODEL/4 = 256 vec4 per row
    xo[i] = wt[i] + wp[i];
}

// ---------------------------------------------------------------------------
// WMMA GEMM: C[M,N] = epilogue(A[M,K] @ B[K,N] + bias[N])
//   RELU / RESID are compile-time; resid adds resid_scale * resid[M,N]
// Workgroup: 256 threads (8 waves), tile 128x128, K-step 32.
// ---------------------------------------------------------------------------
#define LDS_STRIDE 40   // 32 bf16 + 8 pad; 80B rows keep 16B alignment

template <int RELU, int RESID>
__global__ __launch_bounds__(256) void gemm_wmma_kernel(
    const float* __restrict__ A, const float* __restrict__ B,
    const float* __restrict__ bias, const float* __restrict__ resid,
    float resid_scale, float* __restrict__ C, int M, int N, int K) {
    __shared__ __bf16 As[128][LDS_STRIDE];   // [m][k]
    __shared__ __bf16 Bs[128][LDS_STRIDE];   // [n][k]  (transposed tile)

    const int tid  = threadIdx.x;
    const int lane = tid & 31;
    const int wave = tid >> 5;
    const int wm   = wave & 1;     // 0..1  -> 64-row strip
    const int wn   = wave >> 1;    // 0..3  -> 32-col strip
    const int half = lane >> 4;
    const int l16  = lane & 15;

    const int m0 = blockIdx.y * 128;
    const int n0 = blockIdx.x * 128;

    v8f acc[4][2];
#pragma unroll
    for (int mi = 0; mi < 4; ++mi)
#pragma unroll
        for (int ni = 0; ni < 2; ++ni)
#pragma unroll
            for (int r = 0; r < 8; ++r) acc[mi][ni][r] = 0.0f;

    const int ksteps = K >> 5;
    for (int ks = 0; ks < ksteps; ++ks) {
        const int kb = ks << 5;
        __syncthreads();
        // --- A tile 128x32: float4 loads, v4bf (b64) LDS stores ---
#pragma unroll
        for (int i = 0; i < 4; ++i) {
            int u  = tid + 256 * i;
            int r  = u >> 3;               // row in tile
            int c4 = (u & 7) << 2;         // k offset (x4)
            v4f a = *(const v4f*)(A + (size_t)(m0 + r) * K + kb + c4);
            *(v4bf*)(&As[r][c4]) = f2bf4(a);
        }
        // --- B tile 32x128 transposed: 2 float4 loads -> 4 v2bf stores ---
#pragma unroll
        for (int i = 0; i < 2; ++i) {
            int u  = tid + 256 * i;
            int kp = u >> 5;               // k pair index (k = 2*kp)
            int n4 = (u & 31) << 2;        // n offset (x4)
            const float* bp = B + (size_t)(kb + 2 * kp) * N + n0 + n4;
            v4f b0 = *(const v4f*)(bp);
            v4f b1 = *(const v4f*)(bp + N);
#pragma unroll
            for (int j = 0; j < 4; ++j) {
                v2bf p;
                p[0] = f2bf(b0[j]);
                p[1] = f2bf(b1[j]);
                *(v2bf*)(&Bs[n4 + j][2 * kp]) = p;
            }
        }
        // Prefetch next K-tile while this one computes
        if (ks + 1 < ksteps) {
            __builtin_prefetch(A + (size_t)(m0 + (tid >> 1)) * K + kb + 32 +
                                   ((tid & 1) << 4), 0, 0);
            __builtin_prefetch(B + (size_t)(kb + 32 + (tid >> 3)) * N + n0 +
                                   ((tid & 7) << 4), 0, 0);
        }
        __syncthreads();

        v16bf af[4], bfr[2];
#pragma unroll
        for (int mi = 0; mi < 4; ++mi)
            af[mi] = ld_a_frag(&As[wm * 64 + mi * 16 + l16][0], half);
        // B fragment, same k order as A: lane N = l16 fixed, K = half*16 + e contiguous
#pragma unroll
        for (int ni = 0; ni < 2; ++ni)
            bfr[ni] = *(const v16bf*)(&Bs[wn * 32 + ni * 16 + l16][half * 16]);

#pragma unroll
        for (int mi = 0; mi < 4; ++mi)
#pragma unroll
            for (int ni = 0; ni < 2; ++ni)
                acc[mi][ni] = __builtin_amdgcn_wmma_f32_16x16x32_bf16(
                    false, af[mi], false, bfr[ni], (short)0, acc[mi][ni],
                    false, false);
    }

    // Epilogue. C/D layout: VGPR r -> row (r + half*8), col = l16.
#pragma unroll
    for (int ni = 0; ni < 2; ++ni) {
        const int col = n0 + wn * 32 + ni * 16 + l16;
        const float bv = bias[col];
#pragma unroll
        for (int mi = 0; mi < 4; ++mi)
#pragma unroll
            for (int r = 0; r < 8; ++r) {
                const int row = m0 + wm * 64 + mi * 16 + half * 8 + r;
                const size_t off = (size_t)row * N + col;
                float v = acc[mi][ni][r] + bv;
                if (RELU) v = fmaxf(v, 0.0f);
                if (RESID) v = fmaf(resid_scale, resid[off], v);
                C[off] = v;
            }
    }
}

// ---------------------------------------------------------------------------
// Causal attention, flash-style online softmax.
// One wave32 per (b, head, 16-row q-tile); 32 keys per inner iteration.
// qkv: [B,S,3*D] f32 ; y: [B,S,D] f32
// ---------------------------------------------------------------------------
__global__ __launch_bounds__(32) void attn_kernel(
    const float* __restrict__ qkv, float* __restrict__ y) {
    __shared__ __bf16 Qs[16][72];    // [qrow][hd]
    __shared__ __bf16 Ks[32][72];    // [key][hd]     (B-frag for scores)
    __shared__ __bf16 Vt[HDIM][40];  // [hd][key]     (B-frag for P@V)
    __shared__ __bf16 Ps[16][40];    // [qrow][key]   (A-frag bounce)

    const int blk   = blockIdx.x;
    const int qt    = blk & (SEQ / 16 - 1);       // 128 q-tiles
    const int h     = (blk >> 7) & (NHEAD - 1);
    const int b     = blk >> 11;
    const int qbase = qt * 16;

    const int lane = threadIdx.x;
    const int half = lane >> 4;
    const int l16  = lane & 15;

    const size_t rs = 3 * DMODEL;                 // token row stride in qkv
    const float* qp = qkv + (size_t)b * SEQ * rs + (size_t)h * HDIM;
    const float* kp = qp + DMODEL;
    const float* vp = qp + 2 * DMODEL;

    // Stage Q tile 16x64 as bf16 (vec4: 256 units / 32 lanes = 8)
#pragma unroll
    for (int i = 0; i < 8; ++i) {
        int u = lane + 32 * i;
        int r = u >> 4, c4 = (u & 15) << 2;
        v4f q = *(const v4f*)(qp + (size_t)(qbase + r) * rs + c4);
        *(v4bf*)(&Qs[r][c4]) = f2bf4(q);
    }
    __syncthreads();

    v16bf qf[2];
#pragma unroll
    for (int s = 0; s < 2; ++s) qf[s] = ld_a_frag(&Qs[l16][s * 32], half);

    float mrow[8], lrow[8];
    v8f of[4];
#pragma unroll
    for (int r = 0; r < 8; ++r) { mrow[r] = -3.0e38f; lrow[r] = 0.0f; }
#pragma unroll
    for (int j = 0; j < 4; ++j)
#pragma unroll
        for (int r = 0; r < 8; ++r) of[j][r] = 0.0f;

    const float scale = 0.125f;                   // 1/sqrt(64)
    const int nch = (qbase + 16 + 31) >> 5;       // causal chunk count

    for (int ch = 0; ch < nch; ++ch) {
        const int kbase = ch << 5;
        __syncthreads();
        // K chunk [32][64] natural (vec4: 512 units / 32 = 16 per lane)
#pragma unroll
        for (int i = 0; i < 16; ++i) {
            int u = lane + 32 * i;
            int r = u >> 4, c4 = (u & 15) << 2;
            v4f k = *(const v4f*)(kp + (size_t)(kbase + r) * rs + c4);
            *(v4bf*)(&Ks[r][c4]) = f2bf4(k);
        }
        // V chunk transposed [hd][key], key-pairs packed as v2bf
#pragma unroll
        for (int i = 0; i < 8; ++i) {
            int u  = lane + 32 * i;
            int kpair = u >> 4, c4 = (u & 15) << 2;
            const float* vv = vp + (size_t)(kbase + 2 * kpair) * rs + c4;
            v4f x0 = *(const v4f*)(vv);
            v4f x1 = *(const v4f*)(vv + rs);
#pragma unroll
            for (int j = 0; j < 4; ++j) {
                v2bf p;
                p[0] = f2bf(x0[j]);
                p[1] = f2bf(x1[j]);
                *(v2bf*)(&Vt[c4 + j][2 * kpair]) = p;
            }
        }
        __syncthreads();

        // Scores: S_j = Q @ K^T for key groups j of 16; K-dim = hd (2 steps)
        v8f sf[2];
#pragma unroll
        for (int j = 0; j < 2; ++j) {
            v8f z;
#pragma unroll
            for (int r = 0; r < 8; ++r) z[r] = 0.0f;
#pragma unroll
            for (int s = 0; s < 2; ++s) {
                v16bf kf = *(const v16bf*)(&Ks[j * 16 + l16][s * 32 + half * 16]);
                z = __builtin_amdgcn_wmma_f32_16x16x32_bf16(
                    false, qf[s], false, kf, (short)0, z, false, false);
            }
            sf[j] = z;
        }
        // Scale + causal mask (additive -1e10, like the reference)
#pragma unroll
        for (int j = 0; j < 2; ++j)
#pragma unroll
            for (int r = 0; r < 8; ++r) {
                int row = qbase + half * 8 + r;
                int col = kbase + j * 16 + l16;
                sf[j][r] = sf[j][r] * scale + (col > row ? -1.0e10f : 0.0f);
            }
        // Online softmax: DPP16 row reductions (no LDS)
        float newm[8], corr[8];
#pragma unroll
        for (int r = 0; r < 8; ++r) {
            float v = redmax16(fmaxf(sf[0][r], sf[1][r]));
            newm[r] = fmaxf(mrow[r], v);
            corr[r] = __expf(mrow[r] - newm[r]);
        }
#pragma unroll
        for (int j = 0; j < 2; ++j)
#pragma unroll
            for (int r = 0; r < 8; ++r) sf[j][r] = __expf(sf[j][r] - newm[r]);
#pragma unroll
        for (int r = 0; r < 8; ++r) {
            float v = redsum16(sf[0][r] + sf[1][r]);
            lrow[r] = lrow[r] * corr[r] + v;
            mrow[r] = newm[r];
        }
#pragma unroll
        for (int j = 0; j < 4; ++j)
#pragma unroll
            for (int r = 0; r < 8; ++r) of[j][r] *= corr[r];

        // Bounce P (C-layout) through LDS into A-layout
#pragma unroll
        for (int j = 0; j < 2; ++j)
#pragma unroll
            for (int r = 0; r < 8; ++r)
                Ps[half * 8 + r][j * 16 + l16] = f2bf(sf[j][r]);
        __syncthreads();
        v16bf pf = ld_a_frag(&Ps[l16][0], half);

        // O_j += P(16x32) @ V(32x16) over 4 hd-slices
#pragma unroll
        for (int j = 0; j < 4; ++j) {
            v16bf vf = *(const v16bf*)(&Vt[j * 16 + l16][half * 16]);
            of[j] = __builtin_amdgcn_wmma_f32_16x16x32_bf16(
                false, pf, false, vf, (short)0, of[j], false, false);
        }
    }

    // Normalize and write y[b, qrow, h*64 + col]
#pragma unroll
    for (int j = 0; j < 4; ++j)
#pragma unroll
        for (int r = 0; r < 8; ++r) {
            int row = qbase + half * 8 + r;
            int col = j * 16 + l16;
            y[((size_t)b * SEQ + row) * DMODEL + h * HDIM + col] =
                of[j][r] / lrow[r];
        }
}

// ---------------------------------------------------------------------------
// Host-side orchestration
// ---------------------------------------------------------------------------
extern "C" void kernel_launch(void* const* d_in, const int* in_sizes, int n_in,
                              void* d_out, int out_size, void* d_ws,
                              size_t ws_size, hipStream_t stream) {
    (void)in_sizes; (void)n_in; (void)out_size; (void)ws_size;

    const int*   tokens = (const int*)d_in[0];
    const float* wte    = (const float*)d_in[1];
    const float* wpe    = (const float*)d_in[2];
    const float* qkv_w  = (const float*)d_in[3];
    const float* qkv_b  = (const float*)d_in[4];
    const float* o_w    = (const float*)d_in[5];
    const float* o_b    = (const float*)d_in[6];
    const float* f1_w   = (const float*)d_in[7];
    const float* f1_b   = (const float*)d_in[8];
    const float* f2_w   = (const float*)d_in[9];
    const float* f2_b   = (const float*)d_in[10];
    const float* head_w = (const float*)d_in[11];
    const float* head_b = (const float*)d_in[12];
    float* out = (float*)d_out;

    char* ws = (char*)d_ws;
    float* x    = (float*)(ws);                                  // 16 MB
    float* qkv  = (float*)(ws + (size_t)16 * 1024 * 1024);       // 48 MB
    float* y    = (float*)(ws + (size_t)64 * 1024 * 1024);       // 16 MB
    float* hbuf = (float*)(ws + (size_t)80 * 1024 * 1024);       // 64 MB

    embed_kernel<<<MTOK, 256, 0, stream>>>(tokens, wte, wpe, x);

    for (int i = 0; i < 6; ++i) {
        // qkv = x @ qkv_w[i] + qkv_b[i]
        gemm_wmma_kernel<0, 0>
            <<<dim3(3 * DMODEL / 128, MTOK / 128), 256, 0, stream>>>(
                x, qkv_w + (size_t)i * DMODEL * 3 * DMODEL,
                qkv_b + (size_t)i * 3 * DMODEL, nullptr, 0.0f, qkv,
                MTOK, 3 * DMODEL, DMODEL);
        // y = causal_attn(qkv)
        attn_kernel<<<BATCH * NHEAD * (SEQ / 16), 32, 0, stream>>>(qkv, y);
        // x = y @ o_w[i] + o_b[i] + x
        gemm_wmma_kernel<0, 1>
            <<<dim3(DMODEL / 128, MTOK / 128), 256, 0, stream>>>(
                y, o_w + (size_t)i * DMODEL * DMODEL,
                o_b + (size_t)i * DMODEL, x, 1.0f, x, MTOK, DMODEL, DMODEL);
        // h = relu(x @ f1_w[i] + f1_b[i])
        gemm_wmma_kernel<1, 0>
            <<<dim3(FFDIM / 128, MTOK / 128), 256, 0, stream>>>(
                x, f1_w + (size_t)i * DMODEL * FFDIM,
                f1_b + (size_t)i * FFDIM, nullptr, 0.0f, hbuf,
                MTOK, FFDIM, DMODEL);
        // x = (h @ f2_w[i] + f2_b[i]) + 2*x
        gemm_wmma_kernel<0, 1>
            <<<dim3(DMODEL / 128, MTOK / 128), 256, 0, stream>>>(
                hbuf, f2_w + (size_t)i * FFDIM * DMODEL,
                f2_b + (size_t)i * DMODEL, x, 2.0f, x, MTOK, DMODEL, FFDIM);
    }

    // logits = x @ head_w + head_b
    gemm_wmma_kernel<0, 0>
        <<<dim3(VOCAB / 128, MTOK / 128), 256, 0, stream>>>(
            x, head_w, head_b, nullptr, 0.0f, out, MTOK, VOCAB, DMODEL);
}
